// Contrast_loss_point_cloud_70282844832076
// MI455X (gfx1250) — matrix operand order repaired and checked by reading, not compiled, not run on hardware
//
#include <hip/hip_runtime.h>
#include <hip/hip_bf16.h>

// ---------------------------------------------------------------------------
// Supervised contrastive point-cloud loss, CDNA5 (gfx1250) WMMA implementation
// B=8, C=128, N=4096, TEMP=0.1, 16 label classes.
//   loss = mean_b mean_i [ log(sum_{j!=i} e_ij) - log(sum_{j!=i, lab_j==lab_i} e_ij) ]
//   e_ij = exp(<v_i,v_j>/0.1), v = row-normalized features.
//
// Temperature folded into the data: store  w = sqrt(10*log2 e) * v_hat  in
// fp16, so the WMMA f32 output is directly the exp2 argument.
// Each wave owns TWO 16-row strips: every B fragment feeds two WMMAs and the
// two epilogue chains interleave for ILP.
// ---------------------------------------------------------------------------

#define B_DIM   8
#define C_DIM   128
#define N_PTS   4096
#define TILE_J  128
#define LDS_STRIDE 136   // halfs per staged column (272B: bank-conflict-free, 16B aligned)

typedef _Float16 v8h  __attribute__((ext_vector_type(8)));
typedef _Float16 v16h __attribute__((ext_vector_type(16)));
typedef float    v8f  __attribute__((ext_vector_type(8)));

static __device__ __forceinline__ v16h cat16(v8h lo, v8h hi) {
  return __builtin_shufflevector(lo, hi, 0,1,2,3,4,5,6,7,8,9,10,11,12,13,14,15);
}

// ---------------------------------------------------------------------------
// Kernel 1: normalize rows of V = features^T, pre-scale by sqrt(10*log2 e),
// emit fp16 point-major [B][N][C]. Reads coalesced across threads.
// ---------------------------------------------------------------------------
__global__ __launch_bounds__(256) void normalize_kernel(
    const float* __restrict__ feats, _Float16* __restrict__ vh) {
  int idx = blockIdx.x * 256 + threadIdx.x;      // global point: b*N + n
  int b = idx >> 12;
  int n = idx & (N_PTS - 1);
  const float* fb = feats + (size_t)b * C_DIM * N_PTS + n;

  float s = 0.f;
#pragma unroll 8
  for (int c = 0; c < C_DIM; ++c) {
    float x = fb[(size_t)c * N_PTS];
    s += x * x;
  }
  float nrm = __builtin_sqrtf(s);
  nrm = fmaxf(nrm, 1e-12f);
  // sqrt(10 * log2(e)): dot of two scaled vectors == 14.4269504 * cosine
  const float ALPHA = __builtin_sqrtf(14.426950408889634f);
  float scale = ALPHA / nrm;

  _Float16* o = vh + (size_t)idx * C_DIM;
#pragma unroll
  for (int c = 0; c < C_DIM; c += 8) {
    v8h h;
#pragma unroll
    for (int j = 0; j < 8; ++j)
      h[j] = (_Float16)(fb[(size_t)(c + j) * N_PTS] * scale);
    *(v8h*)(o + c) = h;
  }
}

// ---------------------------------------------------------------------------
// Kernel 2: Gram tiles via v_wmma_f32_16x16x32_f16 + fused exp2 / label-masked
// row accumulation. 128 blocks = 8 batches x 16 row-blocks (256 rows each).
// 8 waves/block; each wave owns two 16-row strips (32 rows) and sweeps all
// 4096 columns. Diagonal handled in wave-uniform branches.
// ---------------------------------------------------------------------------
__global__ __launch_bounds__(256) void gram_loss_kernel(
    const _Float16* __restrict__ vh, const int* __restrict__ labels,
    float* __restrict__ partials) {
  __shared__ __align__(16) _Float16 lds_tile[TILE_J * LDS_STRIDE];
  __shared__ int   lds_lab[TILE_J];
  __shared__ float lds_red[16];

  const int tid  = threadIdx.x;
  const int lane = tid & 31;
  const int wave = tid >> 5;
  const int l15  = lane & 15;
  const int hi   = (lane >> 4) & 1;

  const int b  = blockIdx.x >> 4;           // batch
  const int rb = (blockIdx.x & 15) * 256;   // row-block base (256 rows/block)
  const int rw0 = rb + wave * 32;           // strip 0 rows [rw0, rw0+16)
  const int rw1 = rw0 + 16;                 // strip 1 rows [rw1, rw1+16)

  const _Float16* vb = vh + (size_t)b * N_PTS * C_DIM;
  const int*      lb = labels + b * N_PTS;

  // --- A fragments for both strips (f16 A layout) --------------------------
  // lane L (both halves): row M = L&15; halves 0..7 -> K = kk*32 + hi*8 + 0..7
  //                                      halves 8..15 -> K = kk*32+16 + hi*8 + 0..7
  v16h a0[4], a1[4];
#pragma unroll
  for (int kk = 0; kk < 4; ++kk) {
    const v8h* rp0 = (const v8h*)(vb + (size_t)(rw0 + l15) * C_DIM);
    const v8h* rp1 = (const v8h*)(vb + (size_t)(rw1 + l15) * C_DIM);
    a0[kk] = cat16(rp0[kk * 4 + hi], rp0[kk * 4 + hi + 2]);
    a1[kk] = cat16(rp1[kk * 4 + hi], rp1[kk * 4 + hi + 2]);
  }

  // Row labels for each strip's 8 C-fragment rows (per lane half).
  int rlab0[8], rlab1[8];
#pragma unroll
  for (int r = 0; r < 8; ++r) {
    rlab0[r] = lb[rw0 + hi * 8 + r];
    rlab1[r] = lb[rw1 + hi * 8 + r];
  }

  float tot0[8], pos0[8], tot1[8], pos1[8];
#pragma unroll
  for (int r = 0; r < 8; ++r) {
    tot0[r] = 0.f; pos0[r] = 0.f; tot1[r] = 0.f; pos1[r] = 0.f;
  }

  for (int jb = 0; jb < N_PTS; jb += TILE_J) {
    __syncthreads();
    // Stage 128 columns x 128 halfs (32KB) into LDS; 8 x 16B chunks / thread.
    for (int q = tid; q < TILE_J * 16; q += 256) {
      int col = q >> 4;
      int kc  = (q & 15) << 3;
      v8h d = *(const v8h*)(vb + (size_t)(jb + col) * C_DIM + kc);
      *(v8h*)(lds_tile + col * LDS_STRIDE + kc) = d;
    }
    if (tid < TILE_J) lds_lab[tid] = lb[jb + tid];
    __syncthreads();

#pragma unroll 1
    for (int jt = 0; jt < TILE_J / 16; ++jt) {
      // --- B fragments from LDS feed BOTH strips (f16 B layout) ------------
      const _Float16* colp =
          lds_tile + (jt * 16 + l15) * LDS_STRIDE + hi * 8;
      v8f c0 = {}, c1 = {};
#pragma unroll
      for (int kk = 0; kk < 4; ++kk) {
        v8h b0 = *(const v8h*)(colp + kk * 32);
        v8h b1 = *(const v8h*)(colp + kk * 32 + 16);
        v16h bf = cat16(b0, b1);
        c0 = __builtin_amdgcn_wmma_f32_16x16x32_f16(
            false, a0[kk], false, bf, (short)0, c0, false, false);
        c1 = __builtin_amdgcn_wmma_f32_16x16x32_f16(
            false, a1[kk], false, bf, (short)0, c1, false, false);
      }

      // --- fused exp2 + masked accumulation, two interleaved chains --------
      const int clab = lds_lab[jt * 16 + l15];
      float e0[8], e1[8];
#pragma unroll
      for (int r = 0; r < 8; ++r) e0[r] = __builtin_amdgcn_exp2f(c0[r]);
#pragma unroll
      for (int r = 0; r < 8; ++r) e1[r] = __builtin_amdgcn_exp2f(c1[r]);
#pragma unroll
      for (int r = 0; r < 8; ++r) {
        tot0[r] += e0[r];
        pos0[r] += (clab == rlab0[r]) ? e0[r] : 0.f;
        tot1[r] += e1[r];
        pos1[r] += (clab == rlab1[r]) ? e1[r] : 0.f;
      }

      // Diagonal blocks: 16-aligned bases coincide -> wave-uniform, rare.
      const int cb = jb + jt * 16;
      if (cb == rw0) {
#pragma unroll
        for (int r = 0; r < 8; ++r) {
          float d = (l15 == hi * 8 + r) ? e0[r] : 0.f;
          tot0[r] -= d; pos0[r] -= d;
        }
      }
      if (cb == rw1) {
#pragma unroll
        for (int r = 0; r < 8; ++r) {
          float d = (l15 == hi * 8 + r) ? e1[r] : 0.f;
          tot1[r] -= d; pos1[r] -= d;
        }
      }
    }
  }

  // --- reduce the 16 lane-partials per row (stays inside each half-wave) ---
#pragma unroll
  for (int r = 0; r < 8; ++r) {
#pragma unroll
    for (int m = 1; m < 16; m <<= 1) {
      tot0[r] += __shfl_xor(tot0[r], m, 32);
      pos0[r] += __shfl_xor(pos0[r], m, 32);
      tot1[r] += __shfl_xor(tot1[r], m, 32);
      pos1[r] += __shfl_xor(pos1[r], m, 32);
    }
  }

  // lane 0: rows rw0..+7 and rw1..+7; lane 16: rows rw0+8..+15 and rw1+8..+15
  if (l15 == 0) {
    const float LN2 = 0.6931471805599453f;
    float dsum = 0.f;
#pragma unroll
    for (int r = 0; r < 8; ++r) {
      dsum += (__builtin_amdgcn_logf(tot0[r]) -
               __builtin_amdgcn_logf(pos0[r])) * LN2;   // v_log_f32 (log2)
      dsum += (__builtin_amdgcn_logf(tot1[r]) -
               __builtin_amdgcn_logf(pos1[r])) * LN2;
    }
    lds_red[wave * 2 + hi] = dsum;
  }
  __syncthreads();
  if (tid == 0) {
    float s = 0.f;
#pragma unroll
    for (int i = 0; i < 16; ++i) s += lds_red[i];
    partials[blockIdx.x] = s;
  }
}

// ---------------------------------------------------------------------------
// Kernel 3: deterministic tree reduction of block partials -> scalar loss
// ---------------------------------------------------------------------------
__global__ __launch_bounds__(256) void reduce_kernel(
    const float* __restrict__ partials, float* __restrict__ out, int n) {
  __shared__ float s[256];
  int tid = threadIdx.x;
  s[tid] = (tid < n) ? partials[tid] : 0.f;
  __syncthreads();
#pragma unroll
  for (int st = 128; st > 0; st >>= 1) {
    if (tid < st) s[tid] += s[tid + st];
    __syncthreads();
  }
  if (tid == 0) out[0] = s[0] * (1.0f / ((float)B_DIM * (float)N_PTS));
}

// ---------------------------------------------------------------------------
extern "C" void kernel_launch(void* const* d_in, const int* in_sizes, int n_in,
                              void* d_out, int out_size, void* d_ws, size_t ws_size,
                              hipStream_t stream) {
  const float* feats  = (const float*)d_in[0];   // [B, C, N] f32
  const int*   labels = (const int*)d_in[1];     // [B, N]    i32
  float*       out    = (float*)d_out;           // scalar f32

  // workspace: fp16 normalized+scaled V [B][N][C] (8 MB), then block partials
  _Float16* vh = (_Float16*)d_ws;
  float* partials =
      (float*)((char*)d_ws + (size_t)B_DIM * N_PTS * C_DIM * sizeof(_Float16));

  const int n_points = B_DIM * N_PTS;            // 32768
  normalize_kernel<<<n_points / 256, 256, 0, stream>>>(feats, vh);

  const int n_blocks = B_DIM * (N_PTS / 256);    // 128 (32 rows per wave)
  gram_loss_kernel<<<n_blocks, 256, 0, stream>>>(vh, labels, partials);

  reduce_kernel<<<1, 256, 0, stream>>>(partials, out, n_blocks);
}
